// Network_25211458027892
// MI455X (gfx1250) — matrix operand order repaired and checked by reading, not compiled, not run on hardware
//
#include <hip/hip_runtime.h>

// ---------------------------------------------------------------------------
// CDNA5 (gfx1250) implementation of the tensor-field-network reference.
// All dense contractions mapped to v_wmma_f32_16x16x32_f16 (f16 in, f32 acc).
// B-matrices are pre-transposed to [N][K] so every lane's B-fragment is a
// pair of contiguous 16-byte runs (2x b128 loads) per the CDNA5 WMMA lane layout.
// ---------------------------------------------------------------------------

#define BATCH 64
#define NATM  64
#define MULC  64
#define NBAS  16
#define HID   32
#define NPTH  5
#define RMAXF 5.0f

typedef __attribute__((ext_vector_type(16))) _Float16 v16h;
typedef __attribute__((ext_vector_type(8)))  float    v8f;

// ----- WMMA fragment loaders (layouts per CDNA5 ISA 7.12.2, wave32) --------

// A-matrix 16x32 f16 (row-major source): lane&15 = row M; lanes<16 hold
// K 0-7 & 16-23, lanes>=16 hold K 8-15 & 24-31.  -> two 16B contiguous runs.
static __device__ __forceinline__ v16h load_a16(const _Float16* __restrict__ base,
                                                int ld, int lane) {
  const int m  = lane & 15;
  const int kh = (lane >> 4) << 3;  // 0 or 8
  v16h a;
#pragma unroll
  for (int t = 0; t < 16; ++t) {
    const int k = ((t & 8) ? 16 : 0) + kh + (t & 7);
    a[t] = base[m * ld + k];
  }
  return a;
}

// B-matrix 32x16 f16 from a TRANSPOSED [N][K] source: lane&15 = col N;
// lanes<16 hold K 0-7 & 16-23, lanes>=16 hold K 8-15 & 24-31.
static __device__ __forceinline__ v16h load_bT16(const _Float16* __restrict__ base,
                                                 int ldk, int lane) {
  const int n  = lane & 15;
  const int k0 = (lane >> 4) << 3;  // 0 or 8
  v16h b;
#pragma unroll
  for (int t = 0; t < 16; ++t) b[t] = base[n * ldk + ((t & 8) ? 16 : 0) + k0 + (t & 7)];
  return b;
}

static __device__ __forceinline__ v8f wmma_f16(v16h a, v16h b, v8f c) {
  return __builtin_amdgcn_wmma_f32_16x16x32_f16(
      /*neg_a=*/false, a, /*neg_b=*/false, b,
      /*c_mod=*/(short)0, c, /*reuse_a=*/false, /*reuse_b=*/false);
}

// C/D 16x16 f32: VGPR r -> row r (lanes<16) or row r+8 (lanes>=16); col=lane&15.
#define C_ROWBASE(lane) (((lane) >> 4) << 3)
#define C_COL(lane)     ((lane) & 15)

// ---------------------------------------------------------------------------
// prep_kernel: convert layer weights to f16, TRANSPOSED to [N][K] layouts.
//   W1hT  [L][160][32]  (row = p*HID+h, col = k, K padded 16->32 with zeros)
//   W2hT  [L][5][64][32] (row = U, col = k)
//   linshT[L][128][64]   (row = out c, col = U)
//   linvhT[L][64][64]    (row = out v, col = U)
// ---------------------------------------------------------------------------
__global__ void prep_kernel(const float* __restrict__ W1_0, const float* __restrict__ W2_0,
                            const float* __restrict__ LS_0, const float* __restrict__ LV_0,
                            const float* __restrict__ W1_1, const float* __restrict__ W2_1,
                            const float* __restrict__ LS_1, const float* __restrict__ LV_1,
                            _Float16* __restrict__ W1hT, _Float16* __restrict__ W2hT,
                            _Float16* __restrict__ linshT, _Float16* __restrict__ linvhT) {
  const int SEC1 = 2 * 160 * 32;       // 10240
  const int SEC2 = 2 * NPTH * 64 * 32; // 20480
  const int SEC3 = 2 * 128 * 64;       // 16384
  const int SEC4 = 2 * 64 * 64;        // 8192
  const int total = SEC1 + SEC2 + SEC3 + SEC4;
  for (int idx = blockIdx.x * blockDim.x + threadIdx.x; idx < total;
       idx += gridDim.x * blockDim.x) {
    if (idx < SEC1) {
      int l = idx / 5120, rem = idx % 5120;
      int col = rem / 32, k = rem % 32;           // col = p*HID + h
      int p = col / HID, h = col % HID;
      const float* W1 = l ? W1_1 : W1_0;
      float v = (k < NBAS) ? W1[(p * NBAS + k) * HID + h] : 0.0f;
      W1hT[idx] = (_Float16)v;
    } else if (idx < SEC1 + SEC2) {
      int r = idx - SEC1;
      int l = r / 10240, rem = r % 10240;
      int p = rem / 2048, r2 = rem % 2048;
      int U = r2 / 32, k = r2 % 32;
      const float* W2 = l ? W2_1 : W2_0;
      W2hT[r] = (_Float16)W2[(p * HID + k) * MULC + U];
    } else if (idx < SEC1 + SEC2 + SEC3) {
      int r = idx - SEC1 - SEC2;
      int l = r / 8192, rem = r % 8192;
      int c = rem / 64, U = rem % 64;
      const float* LS = l ? LS_1 : LS_0;
      linshT[r] = (_Float16)LS[U * 128 + c];
    } else {
      int r = idx - SEC1 - SEC2 - SEC3;
      int l = r / 4096, rem = r % 4096;
      int v = rem / 64, U = rem % 64;
      const float* LV = l ? LV_1 : LV_0;
      linvhT[r] = (_Float16)LV[U * 64 + v];
    }
  }
}

// ---------------------------------------------------------------------------
// init_kernel: S0 = embed[Z], V0 = 0
// ---------------------------------------------------------------------------
__global__ void init_kernel(const int* __restrict__ Z, const float* __restrict__ embed,
                            float* __restrict__ S0, float* __restrict__ V0) {
  const int idx = blockIdx.x * blockDim.x + threadIdx.x;
  if (idx < BATCH * NATM * MULC) {
    const int bi = idx >> 6, U = idx & 63;
    S0[idx] = embed[Z[bi] * MULC + U];
  }
  if (idx < BATCH * NATM * MULC * 3) V0[idx] = 0.0f;
}

// ---------------------------------------------------------------------------
// edge_kernel: one block per target atom (b,i).  128 threads = 4 wave32.
//   basis[64x16] --wmma--> h[64x160] (relu) --wmma per path--> R_p[64x64]
//   then per-channel CG reductions over j into S_pre / V_pre.
// ---------------------------------------------------------------------------
__global__ __launch_bounds__(128) void edge_kernel(
    const float* __restrict__ pos, const float* __restrict__ mask,
    const float* __restrict__ Sin, const float* __restrict__ Vin,
    const _Float16* __restrict__ W1hT, const float* __restrict__ b1,
    const _Float16* __restrict__ W2hT,
    float* __restrict__ Spre, float* __restrict__ Vpre) {
  __shared__ _Float16 basisA[NATM][32];    // 4 KB  (A matrix, K padded 16->32)
  __shared__ _Float16 hL[NATM][NPTH * HID];// 20 KB
  __shared__ float    Rp[NATM][MULC];      // 16 KB
  __shared__ float    SL[NATM][MULC];      // 16 KB (scaled neighbor scalars)
  __shared__ float    yhatL[NATM][3];
  __shared__ float    scaleL;

  const int bi = blockIdx.x;
  const int b = bi >> 6, i = bi & 63;
  const int tid = threadIdx.x;
  const int lane = tid & 31, wave = tid >> 5;

  if (tid == 0) {
    __builtin_prefetch(W1hT, 0, 3);   // global_prefetch_b8
    __builtin_prefetch(W2hT, 0, 3);
    float s = 0.0f;
    for (int j = 0; j < NATM; ++j) s += mask[b * NATM + j];
    scaleL = rsqrtf(fmaxf(s, 1.0f));
  }
  if (tid < NATM) {
    const int j = tid;
    const float dx = pos[bi * 3 + 0] - pos[(b * NATM + j) * 3 + 0];
    const float dy = pos[bi * 3 + 1] - pos[(b * NATM + j) * 3 + 1];
    const float dz = pos[bi * 3 + 2] - pos[(b * NATM + j) * 3 + 2];
    const float d = sqrtf(dx * dx + dy * dy + dz * dz);
    const float dinv = (d > 1e-9f) ? (1.0f / d) : 0.0f;
    yhatL[j][0] = dx * dinv;
    yhatL[j][1] = dy * dinv;
    yhatL[j][2] = dz * dinv;
    const float em = mask[bi] * mask[b * NATM + j] * ((i == j) ? 0.0f : 1.0f);
    const float step = RMAXF / (float)(NBAS - 1);
    const float gamma = 1.0f / (2.0f * step * step);
#pragma unroll
    for (int k = 0; k < NBAS; ++k) {
      const float t = d - step * (float)k;
      basisA[j][k] = (_Float16)(__expf(-gamma * t * t) * em);
    }
#pragma unroll
    for (int k = NBAS; k < 32; ++k) basisA[j][k] = (_Float16)0.0f;
  }
  __syncthreads();

  // Stage scaled neighbor scalars into LDS (read by the p=0 and p=1 CG passes).
  {
    const float scale = scaleL;
    for (int idx = tid; idx < NATM * MULC; idx += 128) {
      SL[idx >> 6][idx & 63] = Sin[(b * NATM + (idx >> 6)) * MULC + (idx & 63)] * scale;
    }
  }

  // GEMM1: [64x32(pad)] x [32x160] -> h, bias + relu, to LDS as f16.
  for (int tile = wave; tile < 40; tile += 4) {
    const int mt = tile / 10, nt = tile % 10;
    v16h a = load_a16(&basisA[mt * 16][0], 32, lane);
    v16h bf = load_bT16(W1hT + (nt * 16) * 32, 32, lane);
    v8f c = {};
    c = wmma_f16(a, bf, c);
    const int m0 = C_ROWBASE(lane), n = C_COL(lane);
#pragma unroll
    for (int r = 0; r < 8; ++r) {
      const int row = mt * 16 + m0 + r;
      const int col = nt * 16 + n;
      hL[row][col] = (_Float16)fmaxf(c[r] + b1[col], 0.0f);
    }
  }
  __syncthreads();

  float sacc = 0.0f, v0 = 0.0f, v1 = 0.0f, v2 = 0.0f;
  const float scale = scaleL;

  for (int p = 0; p < NPTH; ++p) {
    // GEMM2 path p: [64x32] x [32x64] -> R_p
    for (int tile = wave; tile < 16; tile += 4) {
      const int mt = tile >> 2, ut = tile & 3;
      v16h a = load_a16(&hL[mt * 16][p * HID], NPTH * HID, lane);
      v16h bf = load_bT16(W2hT + (p * MULC + ut * 16) * HID, HID, lane);
      v8f c = {};
      c = wmma_f16(a, bf, c);
      const int m0 = C_ROWBASE(lane), n = C_COL(lane);
#pragma unroll
      for (int r = 0; r < 8; ++r) Rp[mt * 16 + m0 + r][ut * 16 + n] = c[r];
    }
    __syncthreads();

    if (tid < MULC) {
      const int U = tid;
      if (p == 0) {            // R000 : scalar -> scalar
        for (int j = 0; j < NATM; ++j) sacc += Rp[j][U] * SL[j][U];
      } else if (p == 1) {     // R011 : scalar -> vector
        for (int j = 0; j < NATM; ++j) {
          const float rs = Rp[j][U] * SL[j][U];
          v0 += rs * yhatL[j][0];
          v1 += rs * yhatL[j][1];
          v2 += rs * yhatL[j][2];
        }
      } else if (p == 2) {     // R101 : vector -> vector (identity)
        for (int j = 0; j < NATM; ++j) {
          const float r = Rp[j][U] * scale;
          const float* vp = &Vin[((b * NATM + j) * MULC + U) * 3];
          v0 += r * vp[0]; v1 += r * vp[1]; v2 += r * vp[2];
        }
      } else if (p == 3) {     // R110 : vector -> scalar (dot with yhat)
        for (int j = 0; j < NATM; ++j) {
          const float r = Rp[j][U] * scale;
          const float* vp = &Vin[((b * NATM + j) * MULC + U) * 3];
          sacc += r * (yhatL[j][0] * vp[0] + yhatL[j][1] * vp[1] + yhatL[j][2] * vp[2]);
        }
      } else {                 // R111 : vector -> vector (cross = yhat x V)
        for (int j = 0; j < NATM; ++j) {
          const float r = Rp[j][U] * scale;
          const float* vp = &Vin[((b * NATM + j) * MULC + U) * 3];
          const float yx = yhatL[j][0], yy = yhatL[j][1], yz = yhatL[j][2];
          v0 += r * (yy * vp[2] - yz * vp[1]);
          v1 += r * (yz * vp[0] - yx * vp[2]);
          v2 += r * (yx * vp[1] - yy * vp[0]);
        }
      }
    }
    __syncthreads();
  }

  if (tid < MULC) {
    Spre[bi * MULC + tid] = sacc;
    Vpre[(bi * MULC + tid) * 3 + 0] = v0;
    Vpre[(bi * MULC + tid) * 3 + 1] = v1;
    Vpre[(bi * MULC + tid) * 3 + 2] = v2;
  }
}

// ---------------------------------------------------------------------------
// linear_kernel: 16 atoms per block, 128 threads = 4 wave32.
//   s_all = S_pre @ lin_s  (16x64 x 64x128), relu/sigmoid gates
//   V_lin = V_pre @ lin_v  (48x64 x 64x64), gated at store.
// Sout element (bi,U) at Sout[bi*s_stride+U]; Vout (bi,U,a) at
// Vout[bi*v_stride + U*3 + a] (matches both ping-pong bufs and d_out concat).
// ---------------------------------------------------------------------------
__global__ __launch_bounds__(128) void linear_kernel(
    const float* __restrict__ Spre, const float* __restrict__ Vpre,
    const float* __restrict__ mask,
    const _Float16* __restrict__ linshT, const _Float16* __restrict__ linvhT,
    float* __restrict__ Sout, int s_stride,
    float* __restrict__ Vout, int v_stride) {
  __shared__ _Float16 Sp16[16][MULC];       // 2 KB
  __shared__ _Float16 Vp16[48][MULC];       // 6 KB
  __shared__ float    sallL[16][2 * MULC];  // 8 KB
  __shared__ float    gateL[16][MULC];      // 4 KB
  __shared__ float    maskL[16];

  const int a0 = blockIdx.x * 16;
  const int tid = threadIdx.x;
  const int lane = tid & 31, wave = tid >> 5;

  if (tid < 16) maskL[tid] = mask[a0 + tid];
  for (int idx = tid; idx < 16 * MULC; idx += 128) {
    const int row = idx >> 6, U = idx & 63;
    Sp16[row][U] = (_Float16)Spre[(a0 + row) * MULC + U];
  }
  for (int idx = tid; idx < 48 * MULC; idx += 128) {
    const int row = idx >> 6, U = idx & 63;
    const int atom = row / 3, a = row % 3;
    Vp16[row][U] = (_Float16)Vpre[((a0 + atom) * MULC + U) * 3 + a];
  }
  __syncthreads();

  // Phase 1: s_all = S_pre @ lin_s  (K = 64 -> 2 wmma steps)
  for (int nt = wave; nt < 8; nt += 4) {
    v8f c = {};
#pragma unroll
    for (int ks = 0; ks < 2; ++ks) {
      v16h a = load_a16(&Sp16[0][ks * 32], MULC, lane);
      v16h bf = load_bT16(linshT + (nt * 16) * 64 + ks * 32, 64, lane);
      c = wmma_f16(a, bf, c);
    }
    const int m0 = C_ROWBASE(lane), n = C_COL(lane);
#pragma unroll
    for (int r = 0; r < 8; ++r) sallL[m0 + r][nt * 16 + n] = c[r];
  }
  __syncthreads();

  // Gates + scalar output
  for (int idx = tid; idx < 16 * MULC; idx += 128) {
    const int row = idx >> 6, U = idx & 63;
    const float m = maskL[row];
    Sout[(a0 + row) * s_stride + U] = fmaxf(sallL[row][U], 0.0f) * m;
    gateL[row][U] = m / (1.0f + __expf(-sallL[row][MULC + U]));
  }
  __syncthreads();

  // Phase 2: V_lin = V_pre @ lin_v, gated store
  for (int tile = wave; tile < 12; tile += 4) {
    const int mt = tile >> 2, ut = tile & 3;
    v8f c = {};
#pragma unroll
    for (int ks = 0; ks < 2; ++ks) {
      v16h a = load_a16(&Vp16[mt * 16][ks * 32], MULC, lane);
      v16h bf = load_bT16(linvhT + (ut * 16) * 64 + ks * 32, 64, lane);
      c = wmma_f16(a, bf, c);
    }
    const int m0 = C_ROWBASE(lane), n = C_COL(lane);
#pragma unroll
    for (int r = 0; r < 8; ++r) {
      const int row = mt * 16 + m0 + r;     // 0..47
      const int atom = row / 3, a = row % 3;
      const int col = ut * 16 + n;
      Vout[(a0 + atom) * v_stride + col * 3 + a] = c[r] * gateL[atom][col];
    }
  }
}

// ---------------------------------------------------------------------------
// Host-side launcher.  Input order (pytree leaves of setup_inputs()):
//   [0]=Z(int32), [1]=pos, [2]=mask, [3]=embed,
//   then per layer l: W1, W2, b1, lin_s, lin_v.
// Output: [B,N,256] f32 = concat(S, V.reshape(B,N,192)).
// ---------------------------------------------------------------------------
extern "C" void kernel_launch(void* const* d_in, const int* in_sizes, int n_in,
                              void* d_out, int out_size, void* d_ws, size_t ws_size,
                              hipStream_t stream) {
  (void)in_sizes; (void)n_in; (void)out_size; (void)ws_size;

  const int*   Z    = (const int*)d_in[0];
  const float* pos  = (const float*)d_in[1];
  const float* mask = (const float*)d_in[2];
  const float* embed = (const float*)d_in[3];
  const float* W1[2]    = {(const float*)d_in[4], (const float*)d_in[9]};
  const float* W2[2]    = {(const float*)d_in[5], (const float*)d_in[10]};
  const float* b1[2]    = {(const float*)d_in[6], (const float*)d_in[11]};
  const float* lin_s[2] = {(const float*)d_in[7], (const float*)d_in[12]};
  const float* lin_v[2] = {(const float*)d_in[8], (const float*)d_in[13]};

  char* ws = (char*)d_ws;
  const size_t MB = 1024u * 1024u;
  float* Sbuf0 = (float*)(ws + 0 * MB);          // 1 MB
  float* Sbuf1 = (float*)(ws + 1 * MB);          // 1 MB
  float* Vbuf0 = (float*)(ws + 2 * MB);          // 3 MB
  float* Vbuf1 = (float*)(ws + 5 * MB);          // 3 MB
  float* Spre  = (float*)(ws + 8 * MB);          // 1 MB
  float* Vpre  = (float*)(ws + 9 * MB);          // 3 MB
  _Float16* W1hT   = (_Float16*)(ws + 12 * MB);                 // 20 KB
  _Float16* W2hT   = (_Float16*)(ws + 12 * MB + 20 * 1024);     // 40 KB
  _Float16* linshT = (_Float16*)(ws + 12 * MB + 60 * 1024);     // 32 KB
  _Float16* linvhT = (_Float16*)(ws + 12 * MB + 92 * 1024);     // 16 KB

  // 1) weight precision conversion + transpose
  prep_kernel<<<216, 256, 0, stream>>>(W1[0], W2[0], lin_s[0], lin_v[0],
                                       W1[1], W2[1], lin_s[1], lin_v[1],
                                       W1hT, W2hT, linshT, linvhT);
  // 2) embedding / zero-init
  init_kernel<<<(BATCH * NATM * MULC * 3 + 255) / 256, 256, 0, stream>>>(
      Z, embed, Sbuf0, Vbuf0);

  float* Sin = Sbuf0;
  float* Vin = Vbuf0;
  for (int l = 0; l < 2; ++l) {
    edge_kernel<<<BATCH * NATM, 128, 0, stream>>>(
        pos, mask, Sin, Vin,
        W1hT + l * 5120, b1[l], W2hT + l * 10240,
        Spre, Vpre);

    const bool last = (l == 1);
    float* Sout = last ? (float*)d_out : Sbuf1;
    float* Vout = last ? ((float*)d_out + MULC) : Vbuf1;
    const int s_stride = last ? 4 * MULC : MULC;
    const int v_stride = last ? 4 * MULC : 3 * MULC;
    linear_kernel<<<(BATCH * NATM) / 16, 128, 0, stream>>>(
        Spre, Vpre, mask,
        linshT + l * 8192, linvhT + l * 4096,
        Sout, s_stride, Vout, v_stride);

    Sin = Sbuf1;
    Vin = Vbuf1;
  }
}
